// RNNMejorada_35613868818983
// MI455X (gfx1250) — matrix operand order repaired and checked by reading, not compiled, not run on hardware
//
#include <hip/hip_runtime.h>

// ---------------------------------------------------------------------------
// Fused 2-layer RNN scan for MI455X (gfx1250, wave32, WMMA).
//
// Each wave owns a 16-row batch tile and runs the entire S=512 scan of both
// RNN layers with register-resident f16 weight fragments, using
// v_wmma_f32_16x16x32_f16 for the [16,64]x[64,64] hidden GEMMs and a
// wave-private LDS tile to transpose WMMA D-layout -> A-layout between steps.
// No inter-wave synchronization is required (recurrence is batch-parallel).
//
// Round-1 fix: tanhf() pulled in OCML's branchy implementation (saveexec /
// divergent paths between WMMAs, ~40 VALU+SALU per element). Replaced with
// the gfx1250 hardware v_tanh_f32 (TRANS op) when available, else a
// branchless exp2-based tanh.
// ---------------------------------------------------------------------------

typedef __attribute__((ext_vector_type(16))) _Float16 v16h;
typedef __attribute__((ext_vector_type(8)))  _Float16 v8h;
typedef __attribute__((ext_vector_type(8)))  float    v8f;

static __device__ __forceinline__ v8f wmma_f16(v16h a, v16h b, v8f c) {
  // 8 args: (neg_a, A, neg_b, B, c_mod, C, reuse_a, reuse_b)
  return __builtin_amdgcn_wmma_f32_16x16x32_f16(false, a, false, b,
                                                (short)0, c, false, false);
}

// Branchless tanh: hardware V_TANH_F32 on gfx1250, else exp2-based.
static __device__ __forceinline__ float fast_tanh(float v) {
#if __has_builtin(__builtin_amdgcn_tanhf)
  return __builtin_amdgcn_tanhf(v);
#else
  // tanh(x) = 1 - 2/(exp2(x*2*log2e) + 1); clamp keeps exp2 finite.
  float xc = __builtin_fminf(__builtin_fmaxf(v, -9.0f), 9.0f);
  float t  = __builtin_amdgcn_exp2f(xc * 2.8853900817779268f);
  return 1.0f - 2.0f * __builtin_amdgcn_rcpf(t + 1.0f);
#endif
}

#define B_  2048
#define S_  512
#define H_  64
#define ROWS_PER_WAVE   16
#define WAVES_PER_BLOCK 8

__global__ __launch_bounds__(256) void rnn2_fused_kernel(
    const float* __restrict__ x,
    const float* __restrict__ W_ih0,
    const float* __restrict__ W_hh0,
    const float* __restrict__ b_ih0,
    const float* __restrict__ b_hh0,
    const float* __restrict__ W_ih1,
    const float* __restrict__ W_hh1,
    const float* __restrict__ b_ih1,
    const float* __restrict__ b_hh1,
    const float* __restrict__ fc_W,
    const float* __restrict__ fc_b,
    float* __restrict__ out)
{
  const int tid  = threadIdx.x;
  const int lane = tid & 31;
  const int wave = tid >> 5;
  const int l15  = lane & 15;
  const int hi   = lane >> 4;   // 0: lanes 0-15, 1: lanes 16-31
  const int mb   = blockIdx.x * (WAVES_PER_BLOCK * ROWS_PER_WAVE)
                 + wave * ROWS_PER_WAVE;

  // Wave-private staging tiles (16x64 f16, row-major): h1 and h2.
  __shared__ __align__(16) _Float16 stage[WAVES_PER_BLOCK][2][16 * H_];
  _Float16* st1 = &stage[wave][0][0];
  _Float16* st2 = &stage[wave][1][0];

  // --- Preload weight B-fragments. GEMM is h @ W^T, so B(k,n) = W[n,k].
  // B k order used here (contiguous, same as A): lane = n_local + 16*(k_local>>4), half = k_local&15.
  // For lane L: n = ntile*16 + (L&15), k = kf*32 + hi*16 + h  (16 contiguous k).
  auto loadB = [&](const float* __restrict__ W, int ntile, int kf) -> v16h {
    const float* p = W + (ntile * 16 + l15) * H_ + kf * 32 + hi * 16;
    v16h r;
#pragma unroll
    for (int i = 0; i < 4; ++i) {
      float4 q = ((const float4*)p)[i];
      r[4*i+0] = (_Float16)q.x;
      r[4*i+1] = (_Float16)q.y;
      r[4*i+2] = (_Float16)q.z;
      r[4*i+3] = (_Float16)q.w;
    }
    return r;
  };

  v16h Bhh0[4][2], Bih1[4][2], Bhh1[4][2];
#pragma unroll
  for (int n = 0; n < 4; ++n) {
#pragma unroll
    for (int kf = 0; kf < 2; ++kf) {
      Bhh0[n][kf] = loadB(W_hh0, n, kf);
      Bih1[n][kf] = loadB(W_ih1, n, kf);
      Bhh1[n][kf] = loadB(W_hh1, n, kf);
    }
  }

  float bias0[4], bias1[4], wih0[4];
#pragma unroll
  for (int n = 0; n < 4; ++n) {
    int c = n * 16 + l15;
    bias0[n] = b_ih0[c] + b_hh0[c];
    bias1[n] = b_ih1[c] + b_hh1[c];
    wih0[n]  = W_ih0[c];          // W_ih0 is [H, I=1]
  }

  // Hidden states as A-fragments (f16 16x32 per fragment, 2 frags = K 0..63).
  v16h z16 = {};
  v16h h1A[2] = { z16, z16 };
  v16h h2A[2] = { z16, z16 };

  const float* xrow = x + (size_t)(mb + l15) * S_;   // x is [B, S, 1]

#pragma unroll 1
  for (int t = 0; t < S_; ++t) {
    // x(m,t) broadcast: C-tile vgpr r corresponds to row m = hi*8 + r.
    float xv = xrow[t];
    float xm[8];
#pragma unroll
    for (int r = 0; r < 8; ++r) xm[r] = __shfl(xv, hi * 8 + r, 32);

    // ------------------------- layer 0 -------------------------
#pragma unroll
    for (int n = 0; n < 4; ++n) {
      v8f c;
#pragma unroll
      for (int r = 0; r < 8; ++r) c[r] = fmaf(xm[r], wih0[n], bias0[n]);
      c = wmma_f16(h1A[0], Bhh0[n][0], c);
      c = wmma_f16(h1A[1], Bhh0[n][1], c);
      int ncol = n * 16 + l15;
#pragma unroll
      for (int r = 0; r < 8; ++r) {
        float hv = fast_tanh(c[r]);
        st1[(hi * 8 + r) * H_ + ncol] = (_Float16)hv;   // row-major stage
      }
    }
    // Reload h1 as A-fragments: lane L holds row m=L&15; halves 0-15 are
    // K = kf*32 + hi*16 + 0..15 (16 contiguous k, matching B).
#pragma unroll
    for (int kf = 0; kf < 2; ++kf) {
      const v8h* p = (const v8h*)(st1 + l15 * H_ + kf * 32 + hi * 16);
      v8h lo = p[0];
      v8h up = p[1];          // +8 halves
      v16h a;
#pragma unroll
      for (int h = 0; h < 8; ++h) { a[h] = lo[h]; a[8 + h] = up[h]; }
      h1A[kf] = a;
    }

    // ------------------------- layer 1 -------------------------
#pragma unroll
    for (int n = 0; n < 4; ++n) {
      v8f c;
#pragma unroll
      for (int r = 0; r < 8; ++r) c[r] = bias1[n];
      c = wmma_f16(h1A[0], Bih1[n][0], c);
      c = wmma_f16(h1A[1], Bih1[n][1], c);
      c = wmma_f16(h2A[0], Bhh1[n][0], c);
      c = wmma_f16(h2A[1], Bhh1[n][1], c);
      int ncol = n * 16 + l15;
#pragma unroll
      for (int r = 0; r < 8; ++r) {
        float hv = fast_tanh(c[r]);
        st2[(hi * 8 + r) * H_ + ncol] = (_Float16)hv;
      }
    }
#pragma unroll
    for (int kf = 0; kf < 2; ++kf) {
      const v8h* p = (const v8h*)(st2 + l15 * H_ + kf * 32 + hi * 16);
      v8h lo = p[0];
      v8h up = p[1];
      v16h a;
#pragma unroll
      for (int h = 0; h < 8; ++h) { a[h] = lo[h]; a[8 + h] = up[h]; }
      h2A[kf] = a;
    }
  }

  // ---- epilogue: out[b] = h2_last[b,:] . fc_W[0,:] + fc_b  (st2 holds h2) --
  if (lane < 16) {
    float acc = fc_b[0];
    const _Float16* hr = st2 + l15 * H_;
#pragma unroll
    for (int c = 0; c < H_; ++c) acc += (float)hr[c] * fc_W[c];
    out[mb + l15] = acc;
  }
}

extern "C" void kernel_launch(void* const* d_in, const int* in_sizes, int n_in,
                              void* d_out, int out_size, void* d_ws, size_t ws_size,
                              hipStream_t stream) {
  (void)in_sizes; (void)n_in; (void)out_size; (void)d_ws; (void)ws_size;
  const float* x     = (const float*)d_in[0];
  const float* W_ih0 = (const float*)d_in[1];
  const float* W_hh0 = (const float*)d_in[2];
  const float* b_ih0 = (const float*)d_in[3];
  const float* b_hh0 = (const float*)d_in[4];
  const float* W_ih1 = (const float*)d_in[5];
  const float* W_hh1 = (const float*)d_in[6];
  const float* b_ih1 = (const float*)d_in[7];
  const float* b_hh1 = (const float*)d_in[8];
  const float* fc_W  = (const float*)d_in[9];
  const float* fc_b  = (const float*)d_in[10];

  dim3 grid(B_ / (WAVES_PER_BLOCK * ROWS_PER_WAVE));   // 16 blocks
  dim3 block(WAVES_PER_BLOCK * 32);                    // 256 threads = 8 waves
  rnn2_fused_kernel<<<grid, block, 0, stream>>>(
      x, W_ih0, W_hh0, b_ih0, b_hh0,
      W_ih1, W_hh1, b_ih1, b_hh1, fc_W, fc_b, (float*)d_out);
}
